// SwinBlock_57389353009752
// MI455X (gfx1250) — matrix operand order repaired and checked by reading, not compiled, not run on hardware
//
#include <hip/hip_runtime.h>
#include <math.h>

// ---------------------------------------------------------------------------
// Swin block (B=32, C=192, H=W=56, heads=6, ws=7, shift=3, MLP=4) for MI455X
// (gfx1250). All matrix math via v_wmma_f32_16x16x32_f16 (wave32 WMMA),
// f32 accumulate. Convs dominate FLOPs -> implicit-GEMM WMMA with NHWC f16
// activations and [Cout][K] f16 weights (L2-resident on 192MB L2).
// Wave tile 32Mx64N (8 accumulators, 1.5 b128 loads per WMMA); compiler
// handles cross-iteration overlap (no manual rotation -> no spills/movs).
// ---------------------------------------------------------------------------

typedef __attribute__((ext_vector_type(16))) _Float16 v16h;
typedef __attribute__((ext_vector_type(8)))  _Float16 v8h;
typedef __attribute__((ext_vector_type(8)))  float    v8f;

union Frag16 { v16h v; v8h h[2]; };

// A-matrix 16x32 f16 fragment (ISA 7.12.2): lane M = lane&15, hi = lane>>4.
// halfs 0..7  -> K = 8*hi + 0..7 ; halfs 8..15 -> K = 16 + 8*hi + 0..7
__device__ __forceinline__ v16h load_a_frag(const _Float16* rowbase, int hi) {
  Frag16 f;
  f.h[0] = *reinterpret_cast<const v8h*>(rowbase + 8 * hi);
  f.h[1] = *reinterpret_cast<const v8h*>(rowbase + 16 + 8 * hi);
  return f.v;
}

// B-matrix 32x16 f16 fragment: lane N = lane&15, halfs 0..7 -> K = 8*hi + 0..7 ; halfs 8..15 -> K = 16 + 8*hi + 0..7.
// colbase points at Bt[N][K-chunk] (N-major, K contiguous).
__device__ __forceinline__ v16h load_b_frag(const _Float16* colbase, int hi) {
  Frag16 f;
  f.h[0] = *reinterpret_cast<const v8h*>(colbase + 8 * hi);
  f.h[1] = *reinterpret_cast<const v8h*>(colbase + 16 + 8 * hi);
  return f.v;
}

__device__ __forceinline__ v8f wmma_f16(v16h a, v16h b, v8f c) {
  return __builtin_amdgcn_wmma_f32_16x16x32_f16(false, a, false, b,
                                                (short)0, c, false, false);
}

// ------------------------- constants ---------------------------------------
#define BB   32
#define CC   192
#define HH   56
#define WW_  56
#define HW   (HH * WW_)
#define HEADS 6
#define DH   32
#define NWIN 64           // 8x8 windows per image
#define TOK  64           // padded tokens per window (49 real)
#define MROWS (BB * NWIN * TOK)   // 131072 GEMM rows
#define HID  768

// ------------------------- weight converts ---------------------------------
__global__ void k_cvt_f32_f16(const float* __restrict__ src,
                              _Float16* __restrict__ dst, int n) {
  int i = blockIdx.x * 256 + threadIdx.x;
  if (i < n) dst[i] = (_Float16)src[i];
}

// OIHW f32 -> [co][k] f16 with k = (dy*3+dx)*cin + ci
__global__ void k_cvt_conv_w(const float* __restrict__ src,
                             _Float16* __restrict__ dst, int cout, int cin) {
  int i = blockIdx.x * 256 + threadIdx.x;
  int n = cout * cin * 9;
  if (i >= n) return;
  int co  = i / (cin * 9);
  int rem = i - co * (cin * 9);
  int j   = rem / cin;
  int ci  = rem - j * cin;
  dst[i] = (_Float16)src[((size_t)co * cin + ci) * 9 + j];
}

// --------------- LN1 + cyclic shift + window partition ---------------------
// One wave per (window, token). Output A0[row = win*64+tok][192] f16,
// padded tokens (49..63) zeroed.
__global__ void k_ln1_window(const float* __restrict__ x,
                             const float* __restrict__ g,
                             const float* __restrict__ bia,
                             _Float16* __restrict__ A0) {
  int gid  = blockIdx.x * 8 + (threadIdx.x >> 5);   // 0 .. 131071
  int lane = threadIdx.x & 31;
  int win = gid >> 6, tok = gid & 63;
  _Float16* dst = A0 + (size_t)gid * CC;
  if (tok >= 49) {
    #pragma unroll
    for (int j = 0; j < 6; ++j) dst[lane * 6 + j] = (_Float16)0.0f;
    return;
  }
  int b  = win >> 6;
  int wh = (win >> 3) & 7, ww = win & 7;
  int r  = tok / 7, cc = tok - r * 7;
  int hs = wh * 7 + r, ws = ww * 7 + cc;      // shifted-image coords
  int h  = (hs + 3) % HH;                     // roll(-3): shifted[i]=x[(i+3)%56]
  int w  = (ws + 3) % WW_;
  const float* xp = x + (size_t)b * CC * HW + (size_t)h * WW_ + w;
  float vals[6];
  float s = 0.f;
  #pragma unroll
  for (int j = 0; j < 6; ++j) {
    vals[j] = xp[(size_t)(lane * 6 + j) * HW];
    s += vals[j];
  }
  #pragma unroll
  for (int off = 16; off >= 1; off >>= 1) s += __shfl_xor(s, off, 32);
  float mean = s * (1.0f / CC);
  float vs = 0.f;
  #pragma unroll
  for (int j = 0; j < 6; ++j) { float d = vals[j] - mean; vs += d * d; }
  #pragma unroll
  for (int off = 16; off >= 1; off >>= 1) vs += __shfl_xor(vs, off, 32);
  float inv = rsqrtf(vs * (1.0f / CC) + 1e-5f);
  #pragma unroll
  for (int j = 0; j < 6; ++j) {
    int c = lane * 6 + j;
    dst[c] = (_Float16)((vals[j] - mean) * inv * g[c] + bia[c]);
  }
}

// --------------------- generic WMMA GEMM -----------------------------------
// C[m][n] f16 = A[m][k] f16 @ Bt[n][k]^T f16 + bias[n].  K % 32 == 0,
// M % 128 == 0.  Block = 4 waves; wave -> 32(M) x 64(N) (8 accumulators);
// grid = (M/128, N/64).
__global__ void __launch_bounds__(128, 1)
k_gemm(const _Float16* __restrict__ A,
       const _Float16* __restrict__ Bt,
       const float* __restrict__ bias,
       _Float16* __restrict__ C, int M, int N, int K) {
  const int lane = threadIdx.x & 31, wave = threadIdx.x >> 5;
  const int hi = lane >> 4, l16 = lane & 15;
  const int m0 = blockIdx.x * 128 + wave * 32;
  const int n0 = blockIdx.y * 64;
  v8f acc[8] = {};
  const _Float16* arow0 = A + (size_t)(m0 + l16) * K;
  const _Float16* arow1 = A + (size_t)(m0 + 16 + l16) * K;
  const _Float16* bcol[4];
  #pragma unroll
  for (int t = 0; t < 4; ++t)
    bcol[t] = Bt + (size_t)(n0 + t * 16 + l16) * K;

  for (int kc = 0; kc < K; kc += 32) {
    if (kc + 64 < K) __builtin_prefetch(arow0 + kc + 64);  // global_prefetch_b8
    v16h a0 = load_a_frag(arow0 + kc, hi);
    v16h a1 = load_a_frag(arow1 + kc, hi);
    v16h b[4];
    #pragma unroll
    for (int t = 0; t < 4; ++t) b[t] = load_b_frag(bcol[t] + kc, hi);
    #pragma unroll
    for (int t = 0; t < 4; ++t) {
      acc[t]     = wmma_f16(a0, b[t], acc[t]);
      acc[4 + t] = wmma_f16(a1, b[t], acc[4 + t]);
    }
  }
  #pragma unroll
  for (int t = 0; t < 4; ++t) {
    int n = n0 + t * 16 + l16;
    float bv = bias[n];
    #pragma unroll
    for (int i = 0; i < 2; ++i)
      #pragma unroll
      for (int r = 0; r < 8; ++r) {
        int m = m0 + i * 16 + r + 8 * hi;
        C[(size_t)m * N + n] = (_Float16)(acc[i * 4 + t][r] + bv);
      }
  }
}

// --------------------------- attention -------------------------------------
// grid = (2048 windows, 6 heads), block = 128 (4 waves, one per 16-token
// row-slab).  qkv row = win*64+tok, cols: q[h*32+d], k at +192, v at +384.
__global__ void k_attention(const _Float16* __restrict__ qkv,
                            _Float16* __restrict__ attnout) {
  __shared__ _Float16 p_lds[64][64];    // softmaxed scores, row-major
  __shared__ _Float16 vt_lds[32][64];   // V transposed: [d][tok]
  const int win = blockIdx.x, head = blockIdx.y;
  const int tid = threadIdx.x, lane = tid & 31, wave = tid >> 5;
  const int hi = lane >> 4, l16 = lane & 15;
  const _Float16* base = qkv + (size_t)win * TOK * 576 + head * DH;
  const _Float16* Q = base;
  const _Float16* Kp = base + 192;
  const _Float16* Vp = base + 384;

  for (int idx = tid; idx < 32 * 64; idx += 128) {
    int d = idx & 31, t = idx >> 5;
    vt_lds[d][t] = Vp[(size_t)t * 576 + d];
  }

  // scores S = Q(64x32) @ K^T(32x64) * 1/sqrt(32); wave owns 16 rows.
  const int mrow = wave * 16 + l16;
  v16h aq = load_a_frag(Q + (size_t)mrow * 576, hi);
  v8f sacc[4];
  #pragma unroll
  for (int t = 0; t < 4; ++t) {
    v16h bk = load_b_frag(Kp + (size_t)(t * 16 + l16) * 576, hi);
    v8f z = {};
    sacc[t] = wmma_f16(aq, bk, z);
  }

  const float scale = 0.17677669529663687f;   // 32^-0.5
  float p[4][8];
  #pragma unroll
  for (int r = 0; r < 8; ++r) {
    float v[4];
    float mx = -1e30f;
    #pragma unroll
    for (int t = 0; t < 4; ++t) {
      int col = t * 16 + l16;
      v[t] = (col < 49) ? sacc[t][r] * scale : -1e30f;   // mask padded cols
      mx = fmaxf(mx, v[t]);
    }
    #pragma unroll
    for (int off = 8; off >= 1; off >>= 1) mx = fmaxf(mx, __shfl_xor(mx, off, 32));
    float sum = 0.f;
    #pragma unroll
    for (int t = 0; t < 4; ++t) { v[t] = __expf(v[t] - mx); sum += v[t]; }
    #pragma unroll
    for (int off = 8; off >= 1; off >>= 1) sum += __shfl_xor(sum, off, 32);
    float rinv = 1.0f / sum;
    #pragma unroll
    for (int t = 0; t < 4; ++t) p[t][r] = v[t] * rinv;
  }
  #pragma unroll
  for (int t = 0; t < 4; ++t)
    #pragma unroll
    for (int r = 0; r < 8; ++r)
      p_lds[wave * 16 + r + 8 * hi][t * 16 + l16] = (_Float16)p[t][r];
  __syncthreads();

  // out = P(64x64) @ V(64x32); wave owns 16 rows, 2 N-tiles, 2 K-chunks.
  v8f oacc[2] = {};
  #pragma unroll
  for (int kc = 0; kc < 64; kc += 32) {
    v16h ap = load_a_frag(&p_lds[mrow][kc], hi);
    #pragma unroll
    for (int t = 0; t < 2; ++t) {
      v16h bv = load_b_frag(&vt_lds[t * 16 + l16][kc], hi);
      oacc[t] = wmma_f16(ap, bv, oacc[t]);
    }
  }
  _Float16* outp = attnout + (size_t)win * TOK * CC + head * DH;
  #pragma unroll
  for (int t = 0; t < 2; ++t)
    #pragma unroll
    for (int r = 0; r < 8; ++r) {
      int m = wave * 16 + r + 8 * hi;
      outp[(size_t)m * CC + t * 16 + l16] = (_Float16)oacc[t][r];
    }
}

// ------- residual (un-window + un-shift) + LN2 -> x1 (fp32), h0 (NHWC f16) --
__global__ void k_res_ln2(const float* __restrict__ x,
                          const _Float16* __restrict__ projout,
                          const float* __restrict__ g2,
                          const float* __restrict__ b2,
                          float* __restrict__ x1,
                          _Float16* __restrict__ h0) {
  int gid  = blockIdx.x * 8 + (threadIdx.x >> 5);   // 0 .. 100351 pixels
  int lane = threadIdx.x & 31;
  int b = gid / HW, rem = gid - b * HW;
  int h = rem / WW_, w = rem - h * WW_;
  int hs = (h + 53) % HH, ws = (w + 53) % WW_;      // inverse of roll(-3)
  int wh = hs / 7, r = hs - wh * 7;
  int ww = ws / 7, cc = ws - ww * 7;
  size_t row = ((size_t)b * NWIN + wh * 8 + ww) * TOK + r * 7 + cc;
  const _Float16* pp = projout + row * CC;
  const float* xp = x + (size_t)b * CC * HW + (size_t)h * WW_ + w;
  float vals[6];
  float s = 0.f;
  #pragma unroll
  for (int j = 0; j < 6; ++j) {
    int c = lane * 6 + j;
    vals[j] = xp[(size_t)c * HW] + (float)pp[c];
    s += vals[j];
  }
  #pragma unroll
  for (int off = 16; off >= 1; off >>= 1) s += __shfl_xor(s, off, 32);
  float mean = s * (1.0f / CC);
  float vs = 0.f;
  #pragma unroll
  for (int j = 0; j < 6; ++j) { float d = vals[j] - mean; vs += d * d; }
  #pragma unroll
  for (int off = 16; off >= 1; off >>= 1) vs += __shfl_xor(vs, off, 32);
  float inv = rsqrtf(vs * (1.0f / CC) + 1e-5f);
  #pragma unroll
  for (int j = 0; j < 6; ++j) {
    int c = lane * 6 + j;
    x1[(size_t)b * CC * HW + (size_t)c * HW + (size_t)h * WW_ + w] = vals[j];
    h0[(size_t)gid * CC + c] = (_Float16)((vals[j] - mean) * inv * g2[c] + b2[c]);
  }
}

// ----------------- implicit-GEMM 3x3 conv via WMMA -------------------------
// Input NHWC f16 [b][y][x][CIN]; weights Wt [co][k], k = (dy*3+dx)*CIN + ci.
// Linear pixel tiling: block = 128 pixels (4 waves x 32), guarded tail.
// grid = (ceil(3136/128)=25, COUT/64, B).
// GELU_EPI: +bias, exact GELU, write NHWC f16 out16.
// else:     +bias, add residual in-place into io32 (NCHW f32).
template <int CIN, int COUT, bool GELU_EPI>
__global__ void __launch_bounds__(128, 1)
k_conv3x3(const _Float16* __restrict__ in,
          const _Float16* __restrict__ Wt,
          const float* __restrict__ bias,
          _Float16* __restrict__ out16,
          float* __restrict__ io32) {
  const int lane = threadIdx.x & 31, wave = threadIdx.x >> 5;
  const int hi = lane >> 4, l16 = lane & 15;
  const int m0 = blockIdx.x * 128 + wave * 32;   // pixel base within batch
  const int n0 = blockIdx.y * 64;
  const int b  = blockIdx.z;
  constexpr int K = CIN * 9;

  int  py[2], px[2];
  bool pv[2];
  #pragma unroll
  for (int i = 0; i < 2; ++i) {
    int pm = m0 + i * 16 + l16;
    pv[i] = pm < HW;
    int pc = pv[i] ? pm : 0;
    py[i] = pc / WW_;
    px[i] = pc - py[i] * WW_;
  }
  const _Float16* inb = in + (size_t)b * HW * CIN;
  const _Float16* bcol[4];
  #pragma unroll
  for (int t = 0; t < 4; ++t)
    bcol[t] = Wt + (size_t)(n0 + t * 16 + l16) * K;

  v8f acc[8] = {};

  for (int kc = 0; kc < K; kc += 32) {
    const int j   = kc / CIN;                // tap (32 | CIN so chunk in-tap)
    const int ci0 = kc - j * CIN;
    const int dy  = j / 3 - 1, dx = j % 3 - 1;
    if (kc + 64 < K) __builtin_prefetch(bcol[0] + kc + 64);  // global_prefetch_b8
    v16h a[2];
    #pragma unroll
    for (int i = 0; i < 2; ++i) {
      const int sy = py[i] + dy, sx = px[i] + dx;
      v16h av = {};
      if (pv[i] && (unsigned)sy < (unsigned)HH && (unsigned)sx < (unsigned)WW_) {
        const _Float16* ap = inb + ((size_t)sy * WW_ + sx) * CIN + ci0;
        av = load_a_frag(ap, hi);
      }
      a[i] = av;
    }
    v16h bfr[4];
    #pragma unroll
    for (int t = 0; t < 4; ++t) bfr[t] = load_b_frag(bcol[t] + kc, hi);
    #pragma unroll
    for (int t = 0; t < 4; ++t) {
      acc[t]     = wmma_f16(a[0], bfr[t], acc[t]);
      acc[4 + t] = wmma_f16(a[1], bfr[t], acc[4 + t]);
    }
  }

  // --- epilogue ---
  #pragma unroll
  for (int t = 0; t < 4; ++t) {
    int co = n0 + t * 16 + l16;
    float bv = bias[co];
    #pragma unroll
    for (int i = 0; i < 2; ++i)
      #pragma unroll
      for (int r = 0; r < 8; ++r) {
        int pm = m0 + i * 16 + r + 8 * hi;
        if (pm >= HW) continue;
        float v = acc[i * 4 + t][r] + bv;
        if (GELU_EPI) {
          v = 0.5f * v * (1.0f + erff(v * 0.70710678118654752f));
          out16[((size_t)b * HW + pm) * COUT + co] = (_Float16)v;
        } else {
          size_t o = ((size_t)b * COUT + co) * HW + pm;
          io32[o] = io32[o] + v;               // residual: out = x1 + conv2
        }
      }
  }
}

// ---------------------------------------------------------------------------
extern "C" void kernel_launch(void* const* d_in, const int* in_sizes, int n_in,
                              void* d_out, int out_size, void* d_ws,
                              size_t ws_size, hipStream_t stream) {
  const float* x       = (const float*)d_in[0];
  const float* ln1_g   = (const float*)d_in[1];
  const float* ln1_b   = (const float*)d_in[2];
  const float* qkv_w   = (const float*)d_in[3];
  const float* qkv_b   = (const float*)d_in[4];
  const float* proj_w  = (const float*)d_in[5];
  const float* proj_b  = (const float*)d_in[6];
  const float* ln2_g   = (const float*)d_in[7];
  const float* ln2_b   = (const float*)d_in[8];
  const float* conv1_w = (const float*)d_in[9];
  const float* conv1_b = (const float*)d_in[10];
  const float* conv2_w = (const float*)d_in[11];
  const float* conv2_b = (const float*)d_in[12];
  float* out = (float*)d_out;

  // workspace layout (bytes, regions reused across phases; ~250 MB total)
  char* ws = (char*)d_ws;
  const size_t szA0   = (size_t)MROWS * CC * 2;        // 50.3 MB
  const size_t szQkvH = (size_t)BB * HW * HID * 2;     // 154.1 MB (>= qkv 151 MB)
  const size_t szAt   = (size_t)MROWS * CC * 2;        // 50.3 MB (>= h0 38.5 MB)
  _Float16* A0      = (_Float16*)(ws);                 // later: projout
  _Float16* qkvbuf  = (_Float16*)(ws + szA0);          // later: h1 (conv mid)
  _Float16* attnout = (_Float16*)(ws + szA0 + szQkvH); // later: h0 (LN2 out)
  char* wsw = ws + szA0 + szQkvH + szAt;
  _Float16* qkvw16  = (_Float16*)(wsw);                        // 576*192
  _Float16* projw16 = (_Float16*)(wsw + 221184);               // 192*192
  _Float16* w1      = (_Float16*)(wsw + 221184 + 73728);       // 768*1728
  _Float16* w2      = (_Float16*)(wsw + 221184 + 73728 + 2654208); // 192*6912
  _Float16* projout = A0;
  _Float16* h0      = attnout;
  _Float16* h1      = qkvbuf;

  // weights -> f16 (Bt layout [n][k] == W row-major for y = x @ W^T)
  k_cvt_f32_f16<<<(576 * 192 + 255) / 256, 256, 0, stream>>>(qkv_w, qkvw16, 576 * 192);
  k_cvt_f32_f16<<<(192 * 192 + 255) / 256, 256, 0, stream>>>(proj_w, projw16, 192 * 192);
  k_cvt_conv_w<<<(768 * 192 * 9 + 255) / 256, 256, 0, stream>>>(conv1_w, w1, 768, 192);
  k_cvt_conv_w<<<(192 * 768 * 9 + 255) / 256, 256, 0, stream>>>(conv2_w, w2, 192, 768);

  // LN1 + shift + windowize
  k_ln1_window<<<MROWS / 8, 256, 0, stream>>>(x, ln1_g, ln1_b, A0);
  // QKV GEMM: [131072 x 192] @ [192 x 576]
  k_gemm<<<dim3(MROWS / 128, 576 / 64), 128, 0, stream>>>(A0, qkvw16, qkv_b,
                                                          qkvbuf, MROWS, 576, CC);
  // attention per (window, head)
  k_attention<<<dim3(BB * NWIN, HEADS), 128, 0, stream>>>(qkvbuf, attnout);
  // proj GEMM: [131072 x 192] @ [192 x 192]
  k_gemm<<<dim3(MROWS / 128, 192 / 64), 128, 0, stream>>>(attnout, projw16, proj_b,
                                                          projout, MROWS, CC, CC);
  // residual + LN2: x1 -> d_out (NCHW f32), LN2(x1) -> h0 (NHWC f16)
  k_res_ln2<<<(BB * HW) / 8, 256, 0, stream>>>(x, projout, ln2_g, ln2_b, out, h0);
  // conv1 + GELU -> h1 (NHWC f16)
  k_conv3x3<192, 768, true><<<dim3((HW + 127) / 128, 768 / 64, BB), 128, 0, stream>>>(
      h0, w1, conv1_b, h1, nullptr);
  // conv2 + residual -> d_out (reads x1 from d_out, writes final, in place)
  k_conv3x3<768, 192, false><<<dim3((HW + 127) / 128, 192 / 64, BB), 128, 0, stream>>>(
      h1, w2, conv2_b, nullptr, out);
}
